// Dist_Face_head__68942815035842
// MI455X (gfx1250) — matrix operand order repaired and checked by reading, not compiled, not run on hardware
//
#include <hip/hip_runtime.h>
#include <hip/hip_bf16.h>
#include <math.h>

// ---------------------------------------------------------------------------
// ArcFace head loss for MI455X (gfx1250, wave32, WMMA).
//   x[512,512] f32, W[100000,512] f32, targets[512] int -> scalar loss.
// Strategy: bf16 hi/lo split GEMM (3x v_wmma_f32_16x16x32_bf16, f32 accum)
// with fused W-row norms, fixed-max (=S) softmax partials, deterministic
// two-stage reduction. Weights read exactly once from HBM (204.8 MB).
// ---------------------------------------------------------------------------

typedef __bf16 bf16_t;
typedef __bf16 v16bf __attribute__((ext_vector_type(16)));
typedef __bf16 v8bf  __attribute__((ext_vector_type(8)));
typedef float  v8f   __attribute__((ext_vector_type(8)));

#define DIM    512
#define NB     512
#define NCLS   100000
#define CTILE  128
#define NCB    ((NCLS + CTILE - 1) / CTILE)   /* 782 */

#define COS_M_C  0.87758256189037276f
#define SIN_M_C  0.47942553860420301f
#define TH_C   (-0.87758256189037276f)        /* cos(pi - m) = -cos(m) */
#define MM_C     0.23971276930210150f         /* sin(pi - m) * m       */
#define S_C      64.0f

// -------------------- x row-normalize + bf16 hi/lo split -------------------
__global__ __launch_bounds__(256)
void xnorm_kernel(const float* __restrict__ X,
                  bf16_t* __restrict__ Xh, bf16_t* __restrict__ Xl)
{
    __shared__ float red[256];
    const int b = blockIdx.x, t = threadIdx.x;
    const float* xr = X + (size_t)b * DIM;
    float v0 = xr[t];
    float v1 = xr[t + 256];
    red[t] = fmaf(v0, v0, v1 * v1);
    __syncthreads();
    for (int s = 128; s > 0; s >>= 1) {
        if (t < s) red[t] += red[t + s];
        __syncthreads();
    }
    float rn = 1.0f / sqrtf(red[0]);
    float n0 = v0 * rn, n1 = v1 * rn;
    bf16_t h0 = (bf16_t)n0, h1 = (bf16_t)n1;
    size_t o = (size_t)b * DIM + t;
    Xh[o]        = h0;  Xl[o]        = (bf16_t)(n0 - (float)h0);
    Xh[o + 256]  = h1;  Xl[o + 256]  = (bf16_t)(n1 - (float)h1);
}

// ------------------------------- main GEMM ---------------------------------
__global__ __launch_bounds__(256)
void arcface_gemm(const float* __restrict__ W,
                  const bf16_t* __restrict__ Xh,
                  const bf16_t* __restrict__ Xl,
                  const int* __restrict__ targets,
                  float* __restrict__ partial,      // [NB][NCB]
                  float* __restrict__ tgtlog)       // [NB]
{
    __shared__ bf16_t sAh[32 * DIM];     // 32 KB
    __shared__ bf16_t sAl[32 * DIM];     // 32 KB
    __shared__ float  sWave[8][32];
    __shared__ int    sTgt[32];

    const int tid   = threadIdx.x;
    const int lane  = tid & 31;
    const int wave  = tid >> 5;
    const int cb    = blockIdx.x;
    const int mbase = blockIdx.y * 32;
    const int cwave = cb * CTILE + wave * 16;

    // stage normalized-x tile (hi+lo planes) into LDS once per block
    {
        const uint4* gh = reinterpret_cast<const uint4*>(Xh + (size_t)mbase * DIM);
        const uint4* gl = reinterpret_cast<const uint4*>(Xl + (size_t)mbase * DIM);
        uint4* lh = reinterpret_cast<uint4*>(sAh);
        uint4* ll = reinterpret_cast<uint4*>(sAl);
        #pragma unroll
        for (int i = 0; i < (32 * DIM / 8) / 256; ++i) {
            int idx = tid + i * 256;
            lh[idx] = gh[idx];
            ll[idx] = gl[idx];
        }
    }
    if (tid < 32) sTgt[tid] = targets[mbase + tid];
    __syncthreads();

    const int h  = lane >> 4;            // K-half of wave32 per ISA layout
    const int nl = lane & 15;            // A row / B column within fragment
    int wrow = cwave + nl;
    if (wrow > NCLS - 1) wrow = NCLS - 1;            // clamp OOB tail (masked later)
    const float* wbase = W + (size_t)wrow * DIM + h * 8;

    v8f acc0 = {}, acc1 = {};
    float wss = 0.0f;                    // fused ||w_row||^2 accumulation

    for (int k = 0; k < DIM; k += 32) {
        // ---- B fragment: two 8-elem f32 chunks (k+8h, k+16+8h) of one weight row -> bf16 hi/lo
        const float4* wp = reinterpret_cast<const float4*>(wbase + k);
        float4 f0 = wp[0], f1 = wp[1], f2 = wp[4], f3 = wp[5];
        v16bf bh, bl;
        float lss = 0.0f;
        #define CVT(F, I) { float fv_ = (F); bf16_t hv_ = (bf16_t)fv_;        \
                            bh[I] = hv_;                                      \
                            bl[I] = (bf16_t)(fv_ - (float)hv_);               \
                            lss = fmaf(fv_, fv_, lss); }
        CVT(f0.x, 0)  CVT(f0.y, 1)  CVT(f0.z, 2)  CVT(f0.w, 3)
        CVT(f1.x, 4)  CVT(f1.y, 5)  CVT(f1.z, 6)  CVT(f1.w, 7)
        CVT(f2.x, 8)  CVT(f2.y, 9)  CVT(f2.z,10)  CVT(f2.w,11)
        CVT(f3.x,12)  CVT(f3.y,13)  CVT(f3.z,14)  CVT(f3.w,15)
        #undef CVT
        wss += lss;

        // ---- A fragments from LDS: two 8-elem chunks per lane (ISA layout)
        const int ak = k + 8 * h;
        const bf16_t* p0h = sAh + (size_t)nl * DIM + ak;
        const bf16_t* p0l = sAl + (size_t)nl * DIM + ak;
        const bf16_t* p1h = sAh + (size_t)(16 + nl) * DIM + ak;
        const bf16_t* p1l = sAl + (size_t)(16 + nl) * DIM + ak;

        v8bf c0, c1;
        c0 = *(const v8bf*)(p0h); c1 = *(const v8bf*)(p0h + 16);
        v16bf A0h = __builtin_shufflevector(c0, c1, 0,1,2,3,4,5,6,7,8,9,10,11,12,13,14,15);
        c0 = *(const v8bf*)(p0l); c1 = *(const v8bf*)(p0l + 16);
        v16bf A0l = __builtin_shufflevector(c0, c1, 0,1,2,3,4,5,6,7,8,9,10,11,12,13,14,15);
        c0 = *(const v8bf*)(p1h); c1 = *(const v8bf*)(p1h + 16);
        v16bf A1h = __builtin_shufflevector(c0, c1, 0,1,2,3,4,5,6,7,8,9,10,11,12,13,14,15);
        c0 = *(const v8bf*)(p1l); c1 = *(const v8bf*)(p1l + 16);
        v16bf A1l = __builtin_shufflevector(c0, c1, 0,1,2,3,4,5,6,7,8,9,10,11,12,13,14,15);

        // ---- hi/lo split product: hi*hi + hi*lo + lo*hi (3 WMMAs / M-tile)
        acc0 = __builtin_amdgcn_wmma_f32_16x16x32_bf16(false, A0h, false, bh, (short)0, acc0, false, false);
        acc0 = __builtin_amdgcn_wmma_f32_16x16x32_bf16(false, A0h, false, bl, (short)0, acc0, false, false);
        acc0 = __builtin_amdgcn_wmma_f32_16x16x32_bf16(false, A0l, false, bh, (short)0, acc0, false, false);
        acc1 = __builtin_amdgcn_wmma_f32_16x16x32_bf16(false, A1h, false, bh, (short)0, acc1, false, false);
        acc1 = __builtin_amdgcn_wmma_f32_16x16x32_bf16(false, A1h, false, bl, (short)0, acc1, false, false);
        acc1 = __builtin_amdgcn_wmma_f32_16x16x32_bf16(false, A1l, false, bh, (short)0, acc1, false, false);
    }

    // combine the two K-half partial row norms (lane ^ 16)
    wss += __shfl_xor(wss, 16, 32);
    const float rinv = rsqrtf(wss);
    const int  c  = cwave + nl;
    const bool cv = (c < NCLS);

    #pragma unroll
    for (int mt = 0; mt < 2; ++mt) {
        v8f acc = mt ? acc1 : acc0;
        #pragma unroll
        for (int r = 0; r < 8; ++r) {
            const int lm = mt * 16 + 8 * h + r;       // local B-row of this element
            float cosv = acc[r] * rinv;
            float s2   = 1.0f - cosv * cosv;
            s2 = s2 > 0.0f ? s2 : 0.0f;
            float phi  = cosv * COS_M_C - sqrtf(s2) * SIN_M_C;
            phi = (cosv > TH_C) ? phi : (cosv - MM_C);
            const bool isT = cv && (c == sTgt[lm]);
            float logit = S_C * (isT ? phi : cosv);
            if (isT) tgtlog[mbase + lm] = logit;      // unique writer
            // logits <= S always -> softmax with fixed max = S
            float p = cv ? __expf(logit - S_C) : 0.0f;
            p += __shfl_xor(p, 1, 16);
            p += __shfl_xor(p, 2, 16);
            p += __shfl_xor(p, 4, 16);
            p += __shfl_xor(p, 8, 16);
            if (nl == 0) sWave[wave][lm] = p;         // lanes 0 and 16
        }
    }
    __syncthreads();

    // deterministic per-row partial sum-exp for this class tile
    if (tid < 32) {
        float s = 0.0f;
        #pragma unroll
        for (int w = 0; w < 8; ++w) s += sWave[w][tid];
        partial[(size_t)(mbase + tid) * NCB + cb] = s;
    }
}

// ------------------------- final loss reduction ----------------------------
__global__ __launch_bounds__(512)
void arcface_loss(const float* __restrict__ partial,
                  const float* __restrict__ tgtlog,
                  float* __restrict__ out)
{
    __shared__ float red[NB];
    const int b = threadIdx.x;
    const float* pr = partial + (size_t)b * NCB;
    float s = 0.0f;
    for (int i = 0; i < NCB; ++i) s += pr[i];
    float lse = S_C + logf(s);              // logsumexp with fixed max S
    red[b] = lse - tgtlog[b];               // -log p[target]
    __syncthreads();
    for (int st = NB / 2; st > 0; st >>= 1) {
        if (b < st) red[b] += red[b + st];
        __syncthreads();
    }
    if (b == 0) out[0] = red[0] / (float)NB;
}

// ---------------------------------------------------------------------------
extern "C" void kernel_launch(void* const* d_in, const int* in_sizes, int n_in,
                              void* d_out, int out_size, void* d_ws, size_t ws_size,
                              hipStream_t stream)
{
    const float* x   = (const float*)d_in[0];
    const float* w   = (const float*)d_in[1];
    const int*   tgt = (const int*)d_in[2];    // targets (integer per harness convention)
    float* out = (float*)d_out;

    // workspace layout (~2.7 MB total)
    bf16_t* xh      = (bf16_t*)d_ws;                   // 512*512 bf16
    bf16_t* xl      = xh + (size_t)NB * DIM;           // 512*512 bf16
    float*  tgl     = (float*)(xl + (size_t)NB * DIM); // 512 f32
    float*  partial = tgl + NB;                        // 512*782 f32

    xnorm_kernel<<<NB, 256, 0, stream>>>(x, xh, xl);
    dim3 grid(NCB, NB / 32);
    arcface_gemm<<<grid, 256, 0, stream>>>(w, xh, xl, tgt, partial, tgl);
    arcface_loss<<<1, NB, 0, stream>>>(partial, tgl, out);
}
